// SpatialExpert_67723044324016
// MI455X (gfx1250) — matrix operand order repaired and checked by reading, not compiled, not run on hardware
//
#include <hip/hip_runtime.h>
#include <math.h>

// ---------- types ----------
typedef __bf16 bf16_t;
typedef __attribute__((ext_vector_type(16))) __bf16          v16bf;
typedef __attribute__((ext_vector_type(8)))  float           v8f;
typedef __attribute__((ext_vector_type(8)))  unsigned short  us8;
typedef __attribute__((ext_vector_type(16))) unsigned short  us16;

#define GF_GELU 1
#define GF_RES  2

// ---------- helpers ----------
__device__ __forceinline__ bf16_t f2bf(float f) {
  unsigned u = __builtin_bit_cast(unsigned, f);
  unsigned r = u + 0x7FFFu + ((u >> 16) & 1u);   // RNE to bf16
  unsigned short h = (unsigned short)(r >> 16);
  return __builtin_bit_cast(bf16_t, h);
}

__device__ __forceinline__ v8f wmma_bf16(v16bf a, v16bf b, v8f c) {
  return __builtin_amdgcn_wmma_f32_16x16x32_bf16(false, a, false, b, (short)0, c,
                                                 false, false);
}

// A fragment: 16x32 bf16, row-major source with row stride ldk.
// wave32 layout: lanes 0-15 -> M=lane, K0..15 contiguous
//                lanes 16-31 -> M=lane-16, K16..31 contiguous
__device__ __forceinline__ v16bf load_a_frag(const bf16_t* A, int row_base, int ldk, int k0) {
  int lane = threadIdx.x & 31;
  int m = lane & 15, hi = lane >> 4;
  const bf16_t* p = A + (size_t)(row_base + m) * ldk + k0 + hi * 16;
  return __builtin_bit_cast(v16bf, *(const us16*)p);
}

// B fragment: 32x16 bf16 from BT[N][K] (row = output col).
// lanes 0-15: N=lane, K0..15 contiguous; lanes 16-31: N=lane-16, K16..31.
__device__ __forceinline__ v16bf load_b_frag(const bf16_t* BT, int col_base, int ldk, int k0) {
  int lane = threadIdx.x & 31;
  int n = lane & 15, hi = lane >> 4;
  const bf16_t* p = BT + (size_t)(col_base + n) * ldk + k0 + hi * 16;
  return __builtin_bit_cast(v16bf, *(const us16*)p);
}

// ---------- prep: bf16 copy of x + row sq-norms ----------
__global__ void prep_kernel(const float* __restrict__ x, bf16_t* __restrict__ xb,
                            float* __restrict__ sq) {
  int warp = threadIdx.x >> 5, lane = threadIdx.x & 31;
  int row = blockIdx.x * 4 + warp;
  const float* xr = x + (size_t)row * 768;
  bf16_t* xo = xb + (size_t)row * 768;
  float s = 0.f;
  for (int c = lane; c < 768; c += 32) {
    float v = xr[c];
    xo[c] = f2bf(v);
    s += v * v;
  }
  s += __shfl_xor(s, 1, 32);  s += __shfl_xor(s, 2, 32);
  s += __shfl_xor(s, 4, 32);  s += __shfl_xor(s, 8, 32);
  s += __shfl_xor(s, 16, 32);
  if (lane == 0) sq[row] = s;
}

// ---------- weight transpose+convert: W[K,N] f32 -> WT[N,K] bf16 ----------
__global__ void transpose_w_kernel(const float* __restrict__ W, bf16_t* __restrict__ WT,
                                   int K, int N) {
  int idx = blockIdx.x * 256 + threadIdx.x;
  if (idx >= K * N) return;
  int k = idx / N, n = idx - k * N;
  WT[(size_t)n * K + k] = f2bf(W[idx]);
}

// ---------- u = dp_w_pad @ se_w1, v0 = dp_b_pad @ se_w1 ----------
__global__ void dpw_kernel(const float* __restrict__ dp_w, const float* __restrict__ dp_b,
                           const float* __restrict__ se_w1, float* __restrict__ u,
                           float* __restrict__ v0) {
  int c = blockIdx.x * 256 + threadIdx.x;
  if (c >= 768) return;
  float s = 0.f, s2 = 0.f;
  for (int r = 0; r < 192; r++) {
    float w = se_w1[(size_t)r * 768 + c];
    s  += dp_w[r] * w;
    s2 += dp_b[r] * w;
  }
  u[c] = s;  v0[c] = s2;
}

// ---------- gram via WMMA -> mean pairwise distance per row ----------
__global__ __launch_bounds__(128) void gram_meand_kernel(
    const bf16_t* __restrict__ xb, const float* __restrict__ sq,
    float* __restrict__ meand) {
  __shared__ float rowsum[16];
  int warp = threadIdx.x >> 5, lane = threadIdx.x & 31;
  int ncol = lane & 15, hi = lane >> 4;
  int b = blockIdx.x >> 6, rt = blockIdx.x & 63;
  int row_base = rt << 4;
  const bf16_t* X = xb + (size_t)b * 1024 * 768;
  const float* sqb = sq + b * 1024;
  if (threadIdx.x < 16) rowsum[threadIdx.x] = 0.f;
  __syncthreads();
  float racc[8], sqr[8];
#pragma unroll
  for (int r = 0; r < 8; r++) { racc[r] = 0.f; sqr[r] = sqb[row_base + r + hi * 8]; }
#pragma unroll 1
  for (int jc = warp * 64; jc < 1024; jc += 256) {
    v8f a0 = {}, a1 = {}, a2 = {}, a3 = {};
#pragma unroll 1
    for (int k0 = 0; k0 < 768; k0 += 32) {
      v16bf av = load_a_frag(X, row_base, 768, k0);
      v16bf b0 = load_b_frag(X, jc,      768, k0);
      v16bf b1 = load_b_frag(X, jc + 16, 768, k0);
      v16bf b2 = load_b_frag(X, jc + 32, 768, k0);
      v16bf b3 = load_b_frag(X, jc + 48, 768, k0);
      a0 = wmma_bf16(av, b0, a0);  a1 = wmma_bf16(av, b1, a1);
      a2 = wmma_bf16(av, b2, a2);  a3 = wmma_bf16(av, b3, a3);
    }
    v8f accs[4] = {a0, a1, a2, a3};
#pragma unroll
    for (int t = 0; t < 4; t++) {
      float sqc = sqb[jc + t * 16 + ncol];
#pragma unroll
      for (int r = 0; r < 8; r++) {
        float d2 = sqr[r] + sqc - 2.f * accs[t][r];
        racc[r] += sqrtf(fmaxf(d2, 0.f));
      }
    }
  }
#pragma unroll
  for (int r = 0; r < 8; r++) {
    float v = racc[r];
    v += __shfl_xor(v, 1, 32);  v += __shfl_xor(v, 2, 32);
    v += __shfl_xor(v, 4, 32);  v += __shfl_xor(v, 8, 32);
    if (ncol == 0) atomicAdd(&rowsum[r + hi * 8], v);
  }
  __syncthreads();
  if (threadIdx.x < 16)
    meand[b * 1024 + row_base + threadIdx.x] = rowsum[threadIdx.x] * (1.f / 1024.f);
}

// ---------- h1 = gelu(mean_d*u + v0 + se_b1) (GEMM eliminated algebraically) ----------
__global__ void spatial_h1_kernel(const float* __restrict__ meand, const float* __restrict__ u,
                                  const float* __restrict__ v0, const float* __restrict__ se_b1,
                                  bf16_t* __restrict__ h1) {
  int row = blockIdx.x;
  float md = meand[row];
  for (int c = threadIdx.x; c < 768; c += 256) {
    float v = md * u[c] + v0[c] + se_b1[c];
    v = 0.5f * v * (1.f + erff(v * 0.70710678118f));
    h1[(size_t)row * 768 + c] = f2bf(v);
  }
}

// ---------- layernorm: one wave per 768-row, bf16 output ----------
__global__ void layernorm_kernel(const float* __restrict__ x, const float* __restrict__ g,
                                 const float* __restrict__ bta, bf16_t* __restrict__ out) {
  int warp = threadIdx.x >> 5, lane = threadIdx.x & 31;
  int row = blockIdx.x * 4 + warp;
  const float* xr = x + (size_t)row * 768;
  float vals[24];
  float s = 0.f;
#pragma unroll
  for (int i = 0; i < 24; i++) { vals[i] = xr[lane + i * 32]; s += vals[i]; }
  s += __shfl_xor(s, 1, 32);  s += __shfl_xor(s, 2, 32);
  s += __shfl_xor(s, 4, 32);  s += __shfl_xor(s, 8, 32);
  s += __shfl_xor(s, 16, 32);
  float mu = s * (1.f / 768.f);
  float v = 0.f;
#pragma unroll
  for (int i = 0; i < 24; i++) { float d = vals[i] - mu; v += d * d; }
  v += __shfl_xor(v, 1, 32);  v += __shfl_xor(v, 2, 32);
  v += __shfl_xor(v, 4, 32);  v += __shfl_xor(v, 8, 32);
  v += __shfl_xor(v, 16, 32);
  float rstd = rsqrtf(v * (1.f / 768.f) + 1e-5f);
#pragma unroll
  for (int i = 0; i < 24; i++) {
    int c = lane + i * 32;
    out[(size_t)row * 768 + c] = f2bf((vals[i] - mu) * rstd * g[c] + bta[c]);
  }
}

// ---------- generic bf16 WMMA GEMM: 32x64 tile/wave ----------
// C = A[M,K] @ BT[N,K]^T + bias; oscale applied after bias (used to fold 1/sqrt(dh) into Q).
// mode 0: bf16 out row-major [M,N]; mode 1: K head-major [B,H,N,64]; mode 2: V^T [B,H,64,N]
__global__ __launch_bounds__(128) void gemm_bf16_kernel(
    const bf16_t* __restrict__ A, const bf16_t* __restrict__ BT,
    const float* __restrict__ bias, const float* __restrict__ resid,
    float* __restrict__ outf, bf16_t* __restrict__ outb,
    int M, int N, int K, int flags, int mode, float oscale) {
  int warp = threadIdx.x >> 5;
  int wid = blockIdx.x * 4 + warp;
  int tiles_n = N >> 6;
  int tm = wid / tiles_n, tn = wid % tiles_n;
  if (tm >= (M >> 5)) return;
  int row_base = tm << 5, col_base = tn << 6;
  v8f acc[8];
#pragma unroll
  for (int i = 0; i < 8; i++) acc[i] = (v8f){};
#pragma unroll 1
  for (int k0 = 0; k0 < K; k0 += 32) {
    v16bf a0 = load_a_frag(A, row_base,      K, k0);
    v16bf a1 = load_a_frag(A, row_base + 16, K, k0);
    v16bf b0 = load_b_frag(BT, col_base,      K, k0);
    v16bf b1 = load_b_frag(BT, col_base + 16, K, k0);
    v16bf b2 = load_b_frag(BT, col_base + 32, K, k0);
    v16bf b3 = load_b_frag(BT, col_base + 48, K, k0);
    acc[0] = wmma_bf16(a0, b0, acc[0]);  acc[1] = wmma_bf16(a0, b1, acc[1]);
    acc[2] = wmma_bf16(a0, b2, acc[2]);  acc[3] = wmma_bf16(a0, b3, acc[3]);
    acc[4] = wmma_bf16(a1, b0, acc[4]);  acc[5] = wmma_bf16(a1, b1, acc[5]);
    acc[6] = wmma_bf16(a1, b2, acc[6]);  acc[7] = wmma_bf16(a1, b3, acc[7]);
  }
  int lane = threadIdx.x & 31;
  int ncol = lane & 15, hi = lane >> 4;
#pragma unroll
  for (int u = 0; u < 2; u++) {
#pragma unroll
    for (int t = 0; t < 4; t++) {
#pragma unroll
      for (int r = 0; r < 8; r++) {
        int row = row_base + u * 16 + r + hi * 8;
        int col = col_base + t * 16 + ncol;
        float v = acc[u * 4 + t][r];
        if (bias) v += bias[col];
        v *= oscale;
        if (flags & GF_GELU) v = 0.5f * v * (1.f + erff(v * 0.70710678118f));
        if (flags & GF_RES)  v += resid[(size_t)row * N + col];
        if (outf) outf[(size_t)row * N + col] = v;
        if (outb) {
          size_t idx;
          if (mode == 0) {
            idx = (size_t)row * N + col;
          } else {
            int b = row >> 10, tok = row & 1023;
            int h = col >> 6, kd = col & 63;
            if (mode == 1) idx = (((size_t)(b * 12 + h)) * 1024 + tok) * 64 + kd;
            else           idx = (((size_t)(b * 12 + h)) * 64 + kd) * 1024 + tok;
          }
          outb[idx] = f2bf(v);
        }
      }
    }
  }
}

// ---------- flash attention: wave per (b,h,16 rows), online softmax ----------
// Q already carries the 1/sqrt(dh) scale (folded into the Q-projection GEMM).
__global__ __launch_bounds__(128) void attention_kernel(
    const bf16_t* __restrict__ q, const bf16_t* __restrict__ khm,
    const bf16_t* __restrict__ vt, bf16_t* __restrict__ o) {
  __shared__ __align__(64) bf16_t pbuf[4][16 * 32];
  int warp = threadIdx.x >> 5, lane = threadIdx.x & 31;
  int ncol = lane & 15, hi = lane >> 4;
  int gid = blockIdx.x * 4 + warp;        // 0..6143
  int rt = gid & 63, bh = gid >> 6;       // bh 0..95
  int b = bh / 12, h = bh - b * 12;
  int row_base = rt << 4;
  const bf16_t* Q  = q   + (size_t)b * 1024 * 768 + h * 64;
  const bf16_t* Kh = khm + (size_t)bh * 1024 * 64;    // [tok,64]
  const bf16_t* Vh = vt  + (size_t)bh * 64 * 1024;    // [64,tok]
  bf16_t* P = pbuf[warp];
  // Q fragments are invariant over the j loop: hoist them.
  v16bf qa0 = load_a_frag(Q, row_base, 768, 0);
  v16bf qa1 = load_a_frag(Q, row_base, 768, 32);
  float m[8], l[8];
#pragma unroll
  for (int r = 0; r < 8; r++) { m[r] = -1e30f; l[r] = 0.f; }
  v8f o0 = {}, o1 = {}, o2 = {}, o3 = {};
#pragma unroll 1
  for (int j = 0; j < 1024; j += 32) {
    v8f s0 = {}, s1 = {};
    {
      v16bf b00 = load_b_frag(Kh, j,      64, 0);
      v16bf b10 = load_b_frag(Kh, j + 16, 64, 0);
      v16bf b01 = load_b_frag(Kh, j,      64, 32);
      v16bf b11 = load_b_frag(Kh, j + 16, 64, 32);
      s0 = wmma_bf16(qa0, b00, s0);
      s1 = wmma_bf16(qa0, b10, s1);
      s0 = wmma_bf16(qa1, b01, s0);
      s1 = wmma_bf16(qa1, b11, s1);
    }
#pragma unroll
    for (int r = 0; r < 8; r++) {
      float x0 = s0[r], x1 = s1[r];
      float tmax = fmaxf(x0, x1);
      tmax = fmaxf(tmax, __shfl_xor(tmax, 1, 32));
      tmax = fmaxf(tmax, __shfl_xor(tmax, 2, 32));
      tmax = fmaxf(tmax, __shfl_xor(tmax, 4, 32));
      tmax = fmaxf(tmax, __shfl_xor(tmax, 8, 32));
      float mn = fmaxf(m[r], tmax);
      float corr = expf(m[r] - mn);
      float p0 = expf(x0 - mn), p1 = expf(x1 - mn);
      float rs = p0 + p1;
      rs += __shfl_xor(rs, 1, 32);  rs += __shfl_xor(rs, 2, 32);
      rs += __shfl_xor(rs, 4, 32);  rs += __shfl_xor(rs, 8, 32);
      l[r] = l[r] * corr + rs;
      m[r] = mn;
      o0[r] *= corr;  o1[r] *= corr;  o2[r] *= corr;  o3[r] *= corr;
      int prow = r + hi * 8;
      P[prow * 32 + ncol]      = f2bf(p0);
      P[prow * 32 + 16 + ncol] = f2bf(p1);
    }
    asm volatile("s_wait_dscnt 0" ::: "memory");  // P stores (same wave, in-order DS) before reads
    v16bf pa = load_a_frag(P, 0, 32, 0);
    v16bf v0f = load_b_frag(Vh, 0,  1024, j);
    v16bf v1f = load_b_frag(Vh, 16, 1024, j);
    v16bf v2f = load_b_frag(Vh, 32, 1024, j);
    v16bf v3f = load_b_frag(Vh, 48, 1024, j);
    o0 = wmma_bf16(pa, v0f, o0);  o1 = wmma_bf16(pa, v1f, o1);
    o2 = wmma_bf16(pa, v2f, o2);  o3 = wmma_bf16(pa, v3f, o3);
  }
  v8f oo[4] = {o0, o1, o2, o3};
#pragma unroll
  for (int t = 0; t < 4; t++) {
#pragma unroll
    for (int r = 0; r < 8; r++) {
      int row = row_base + r + hi * 8;
      int col = t * 16 + ncol;
      float val = oo[t][r] / l[r];
      o[((size_t)(b * 1024 + row)) * 768 + h * 64 + col] = f2bf(val);
    }
  }
}

// ---------- host orchestration ----------
extern "C" void kernel_launch(void* const* d_in, const int* in_sizes, int n_in,
                              void* d_out, int out_size, void* d_ws, size_t ws_size,
                              hipStream_t stream) {
  (void)in_sizes; (void)n_in; (void)out_size; (void)ws_size;
  const float* x     = (const float*)d_in[0];
  const float* dp_w  = (const float*)d_in[1];
  const float* dp_b  = (const float*)d_in[2];
  const float* se_w1 = (const float*)d_in[3];
  const float* se_b1 = (const float*)d_in[4];
  const float* se_w2 = (const float*)d_in[5];
  const float* se_b2 = (const float*)d_in[6];
  const float* wq = (const float*)d_in[7];   const float* bq = (const float*)d_in[8];
  const float* wk = (const float*)d_in[9];   const float* bk = (const float*)d_in[10];
  const float* wv = (const float*)d_in[11];  const float* bv = (const float*)d_in[12];
  const float* wo = (const float*)d_in[13];  const float* bo = (const float*)d_in[14];
  const float* gw1 = (const float*)d_in[15]; const float* gb1 = (const float*)d_in[16];
  const float* gw2 = (const float*)d_in[17]; const float* gb2 = (const float*)d_in[18];
  const float* n1g = (const float*)d_in[19]; const float* n1b = (const float*)d_in[20];
  const float* n2g = (const float*)d_in[21]; const float* n2b = (const float*)d_in[22];
  float* out = (float*)d_out;

  const size_t R = 8192, D = 768, DFF = 3072;
  char* ws = (char*)d_ws;
  size_t off = 0;
  auto carve = [&](size_t bytes) -> char* {
    char* p = ws + off;
    off += (bytes + 255) & ~(size_t)255;
    return p;
  };
  bf16_t* xb   = (bf16_t*)carve(R * D * 2);     // x bf16; reused as LN output (xn)
  bf16_t* h1   = (bf16_t*)carve(R * D * 2);     // spatial hidden; reused as attn output
  bf16_t* qb   = (bf16_t*)carve(R * D * 2);
  bf16_t* khm  = (bf16_t*)carve(R * D * 2);     // K head-major
  bf16_t* vtb  = (bf16_t*)carve(R * D * 2);     // V transposed per head
  bf16_t* hff  = (bf16_t*)carve(R * DFF * 2);
  float*  x1   = (float*)carve(R * D * 4);
  float*  x2   = (float*)carve(R * D * 4);
  float*  sq   = (float*)carve(R * 4);
  float*  meand= (float*)carve(R * 4);
  float*  u    = (float*)carve(D * 4);
  float*  v0   = (float*)carve(D * 4);
  bf16_t* se_w2_t = (bf16_t*)carve(D * D * 2);
  bf16_t* wq_t    = (bf16_t*)carve(D * D * 2);
  bf16_t* wk_t    = (bf16_t*)carve(D * D * 2);
  bf16_t* wv_t    = (bf16_t*)carve(D * D * 2);
  bf16_t* wo_t    = (bf16_t*)carve(D * D * 2);
  bf16_t* gw1_t   = (bf16_t*)carve(D * DFF * 2);
  bf16_t* gw2_t   = (bf16_t*)carve(DFF * D * 2);

  auto gblocks = [](int M, int N) { return ((M / 32) * (N / 64) + 3) / 4; };

  prep_kernel<<<2048, 128, 0, stream>>>(x, xb, sq);
  transpose_w_kernel<<<(768 * 768 + 255) / 256, 256, 0, stream>>>(se_w2, se_w2_t, 768, 768);
  transpose_w_kernel<<<(768 * 768 + 255) / 256, 256, 0, stream>>>(wq, wq_t, 768, 768);
  transpose_w_kernel<<<(768 * 768 + 255) / 256, 256, 0, stream>>>(wk, wk_t, 768, 768);
  transpose_w_kernel<<<(768 * 768 + 255) / 256, 256, 0, stream>>>(wv, wv_t, 768, 768);
  transpose_w_kernel<<<(768 * 768 + 255) / 256, 256, 0, stream>>>(wo, wo_t, 768, 768);
  transpose_w_kernel<<<(768 * 3072 + 255) / 256, 256, 0, stream>>>(gw1, gw1_t, 768, 3072);
  transpose_w_kernel<<<(3072 * 768 + 255) / 256, 256, 0, stream>>>(gw2, gw2_t, 3072, 768);
  dpw_kernel<<<3, 256, 0, stream>>>(dp_w, dp_b, se_w1, u, v0);

  gram_meand_kernel<<<512, 128, 0, stream>>>(xb, sq, meand);
  spatial_h1_kernel<<<8192, 256, 0, stream>>>(meand, u, v0, se_b1, h1);
  // x1 = x + h1 @ se_w2 + se_b2
  gemm_bf16_kernel<<<gblocks(8192, 768), 128, 0, stream>>>(
      h1, se_w2_t, se_b2, x, x1, nullptr, 8192, 768, 768, GF_RES, 0, 1.0f);

  layernorm_kernel<<<2048, 128, 0, stream>>>(x1, n1g, n1b, xb);
  // Q projection with softmax scale folded in (1/sqrt(64))
  gemm_bf16_kernel<<<gblocks(8192, 768), 128, 0, stream>>>(
      xb, wq_t, bq, nullptr, nullptr, qb, 8192, 768, 768, 0, 0, 0.125f);
  gemm_bf16_kernel<<<gblocks(8192, 768), 128, 0, stream>>>(
      xb, wk_t, bk, nullptr, nullptr, khm, 8192, 768, 768, 0, 1, 1.0f);
  gemm_bf16_kernel<<<gblocks(8192, 768), 128, 0, stream>>>(
      xb, wv_t, bv, nullptr, nullptr, vtb, 8192, 768, 768, 0, 2, 1.0f);

  attention_kernel<<<1536, 128, 0, stream>>>(qb, khm, vtb, h1);
  // x2 = x1 + attn_o @ wo + bo
  gemm_bf16_kernel<<<gblocks(8192, 768), 128, 0, stream>>>(
      h1, wo_t, bo, x1, x2, nullptr, 8192, 768, 768, GF_RES, 0, 1.0f);

  layernorm_kernel<<<2048, 128, 0, stream>>>(x2, n2g, n2b, xb);
  gemm_bf16_kernel<<<gblocks(8192, 3072), 128, 0, stream>>>(
      xb, gw1_t, gb1, nullptr, nullptr, hff, 8192, 3072, 768, GF_GELU, 0, 1.0f);
  // out = x2 + hff @ gw2 + gb2
  gemm_bf16_kernel<<<gblocks(8192, 768), 128, 0, stream>>>(
      hff, gw2_t, gb2, x2, out, nullptr, 8192, 768, 3072, GF_RES, 0, 1.0f);
}
